// CTMBlock_66065186947569
// MI455X (gfx1250) — matrix operand order repaired and checked by reading, not compiled, not run on hardware
//
#include <hip/hip_runtime.h>
#include <hip/hip_bf16.h>
#include <math.h>

// ---------------- problem constants (from reference) ----------------
constexpr int BDIM  = 2;
constexpr int TDIM  = 2048;
constexpr int DIM   = 768;
constexpr int BT    = BDIM * TDIM;       // 4096 rows
constexpr int TWO_D = 2 * DIM;           // 1536
constexpr int KIT   = 8;                 // ctm iterations
constexpr int MTR   = 8;                 // memory length
constexpr int HID   = 16;                // memory hidden
constexpr int NS    = 512;               // n_synch

// ---------------- WMMA vector types (gfx1250, wave32) ----------------
typedef __attribute__((ext_vector_type(16))) _Float16 v16h;
typedef __attribute__((ext_vector_type(8)))  _Float16 v8h;
typedef __attribute__((ext_vector_type(8)))  float    v8f;

union V16 { v16h v; v8h h[2]; };

__device__ __forceinline__ float sigmoidf(float x) {
    return 1.0f / (1.0f + expf(-x));
}

// 32-bit LDS offset of a generic pointer to __shared__ memory
__device__ __forceinline__ uint32_t lds_off(const void* p) {
    return (uint32_t)(uintptr_t)(__attribute__((address_space(3))) const void*)p;
}

// gfx1250 async global->LDS copy, 16B per lane, tracked by ASYNCcnt
__device__ __forceinline__ void async_b128(uint32_t lds_addr, const void* gaddr) {
    asm volatile("global_load_async_to_lds_b128 %0, %1, off"
                 :: "v"(lds_addr), "v"((unsigned long long)(uintptr_t)gaddr)
                 : "memory");
}

// =====================================================================
// Generic WMMA GEMM:  C (M x N, f32) = A (M x K, f16 row-major)
//                                    @ W^T, W given as (N x K, f16 row-major)
// 256 threads = 8 waves. Macro-tile 256(M) x 128(N), K-step 32.
// LDS double-buffered, filled with GLOBAL_LOAD_ASYNC_TO_LDS_B128 and
// pipelined on ASYNCcnt (branch-free steady state + peeled epilogue).
// Wave grid 4(M) x 2(N); each wave computes a 4x4 block of 16x16 tiles
// (16 v_wmma per K-step per wave).
// M % 256 == 0, N % 128 == 0, K % 64 == 0 (true for 4096/1536/768/512).
// =====================================================================
__global__ __launch_bounds__(256) void wmma_gemm_nt(
    const _Float16* __restrict__ A, const _Float16* __restrict__ W,
    float* __restrict__ C, int M, int N, int Kd) {
    __shared__ _Float16 As[2][256 * 32];   // 2 x 16 KB
    __shared__ _Float16 Bs[2][128 * 32];   // 2 x  8 KB

    const int tid  = threadIdx.x;
    const int lane = tid & 31;
    const int wave = tid >> 5;
    const int wm   = wave & 3;       // wave row: 0..3 -> 64-row strip
    const int wn   = wave >> 2;      // wave col: 0..1 -> 64-col strip
    const int lm   = lane & 15;      // lane within 16
    const int hi   = lane >> 4;      // half-wave select

    const int blockN = blockIdx.x * 128;
    const int blockM = blockIdx.y * 256;

    // per-tile async issue: 1024 A-chunks + 512 B-chunks of 16B, 6/thread
    auto issue_tile = [&](int k0, int stage) {
#pragma unroll
        for (int it = 0; it < 6; ++it) {
            int i = tid + it * 256;          // 0..1535, branch uniform per it
            if (i < 1024) {
                int row = i >> 2, cg = i & 3;
                async_b128(lds_off(&As[stage][row * 32 + cg * 8]),
                           &A[(size_t)(blockM + row) * Kd + k0 + cg * 8]);
            } else {
                int j = i - 1024;
                int row = j >> 2, cg = j & 3;
                async_b128(lds_off(&Bs[stage][row * 32 + cg * 8]),
                           &W[(size_t)(blockN + row) * Kd + k0 + cg * 8]);
            }
        }
    };

    v8f acc[4][4];
#pragma unroll
    for (int mt = 0; mt < 4; ++mt)
#pragma unroll
        for (int nt = 0; nt < 4; ++nt) {
            v8f z = {};
            acc[mt][nt] = z;
        }

    // fragment loads follow ISA 7.12.2 wave32 layouts:
    //   A: lane(m=lm, kb=8*hi),  element j -> k = kb + 16*(j>>3) + (j&7)
    //   B: lane(n=lm, kb=8*hi),  element j -> k = kb + 16*(j>>3) + (j&7)  (b[j] = W[n][k])
    auto compute_tile = [&](int stage) {
        V16 afr[4];
#pragma unroll
        for (int mt = 0; mt < 4; ++mt) {
            int r = wm * 64 + mt * 16 + lm;
            afr[mt].h[0] = *(const v8h*)&As[stage][r * 32 + hi * 8];
            afr[mt].h[1] = *(const v8h*)&As[stage][r * 32 + hi * 8 + 16];
        }
        V16 bfr[4];
#pragma unroll
        for (int nt = 0; nt < 4; ++nt) {
            int n = wn * 64 + nt * 16 + lm;
            bfr[nt].h[0] = *(const v8h*)&Bs[stage][n * 32 + hi * 8];
            bfr[nt].h[1] = *(const v8h*)&Bs[stage][n * 32 + hi * 8 + 16];
        }
#pragma unroll
        for (int mt = 0; mt < 4; ++mt)
#pragma unroll
            for (int nt = 0; nt < 4; ++nt)
                acc[mt][nt] = __builtin_amdgcn_wmma_f32_16x16x32_f16(
                    false, afr[mt].v, false, bfr[nt].v,
                    (short)0, acc[mt][nt], false, false);
    };

    const int NT = Kd / 32;

    // prologue: tile 0 in flight
    issue_tile(0, 0);

    // steady state: branch-free; always one tile in flight
    for (int t = 0; t < NT - 1; ++t) {
        const int stage = t & 1;
        issue_tile((t + 1) * 32, stage ^ 1);
        // drain tile t (6 older async ops), leave tile t+1 in flight
        asm volatile("s_wait_asynccnt 0x6" ::: "memory");
        __syncthreads();            // all waves' tile-t LDS writes visible
        compute_tile(stage);
        __syncthreads();            // before buffer[stage] refill (tile t+2)
    }

    // epilogue: last tile
    asm volatile("s_wait_asynccnt 0x0" ::: "memory");
    __syncthreads();
    compute_tile((NT - 1) & 1);

    // --- store: row = 8*hi + v, col = lm within each 16x16 tile ---
#pragma unroll
    for (int mt = 0; mt < 4; ++mt)
#pragma unroll
        for (int nt = 0; nt < 4; ++nt) {
            int row0 = blockM + wm * 64 + mt * 16 + hi * 8;
            int col  = blockN + wn * 64 + nt * 16 + lm;
#pragma unroll
            for (int v = 0; v < 8; ++v)
                C[(size_t)(row0 + v) * N + col] = acc[mt][nt][v];
        }
}

// =====================================================================
// Prep kernels
// =====================================================================
__global__ void cvt_f32_f16(const float* __restrict__ in,
                            _Float16* __restrict__ out, int n) {
    int g = blockIdx.x * blockDim.x + threadIdx.x;
    if (g < n) out[g] = (_Float16)in[g];
}

__global__ void init_pre(const float* __restrict__ x,
                         const float* __restrict__ start_state,
                         _Float16* __restrict__ pre) {
    int g = blockIdx.x * blockDim.x + threadIdx.x;   // BT*TWO_D
    if (g >= BT * TWO_D) return;
    int b = g / TWO_D, c = g - b * TWO_D;
    float v = (c < DIM) ? x[b * DIM + c] : start_state[c - DIM];
    pre[g] = (_Float16)v;
}

// =====================================================================
// GLU + LayerNorm over D=768 per row; writes trace-history slot k.
// One 256-thread block per row; 3 elements per thread.
// =====================================================================
__global__ __launch_bounds__(256) void glu_ln(
    const float* __restrict__ act, const float* __restrict__ g,
    const float* __restrict__ bln, float* __restrict__ nslot) {
    __shared__ float red[256];
    __shared__ float s_mu, s_rstd;
    const int b   = blockIdx.x;
    const int tid = threadIdx.x;
    const size_t base = (size_t)b * TWO_D;

    float lv[3];
    float sum = 0.0f;
#pragma unroll
    for (int j = 0; j < 3; ++j) {
        int i = tid + j * 256;
        float v  = act[base + i];
        float gt = act[base + DIM + i];
        lv[j] = v * sigmoidf(gt);
        sum += lv[j];
    }
    red[tid] = sum;
    __syncthreads();
    for (int s = 128; s > 0; s >>= 1) {
        if (tid < s) red[tid] += red[tid + s];
        __syncthreads();
    }
    if (tid == 0) s_mu = red[0] * (1.0f / DIM);
    __syncthreads();
    float mu = s_mu;
    float sq = 0.0f;
#pragma unroll
    for (int j = 0; j < 3; ++j) {
        float d = lv[j] - mu;
        sq += d * d;
    }
    red[tid] = sq;
    __syncthreads();
    for (int s = 128; s > 0; s >>= 1) {
        if (tid < s) red[tid] += red[tid + s];
        __syncthreads();
    }
    if (tid == 0) s_rstd = rsqrtf(red[0] * (1.0f / DIM) + 1e-5f);
    __syncthreads();
    float rstd = s_rstd;
#pragma unroll
    for (int j = 0; j < 3; ++j) {
        int i = tid + j * 256;
        nslot[(size_t)b * DIM + i] = (lv[j] - mu) * rstd * g[i] + bln[i];
    }
}

// =====================================================================
// Per-neuron memory MLP (NLM): trace(8) -> 32 -GLU-> 16 -> 2 -GLU-> 1.
// Trace ring: logical m < 7-k -> start_trace[n][m+k+1], else nhist slot.
// Writes state f32 and its f16 copy into pre[:, D:2D].
// =====================================================================
__global__ __launch_bounds__(256) void nlm_step(
    const float* __restrict__ nhist,        // KIT slots of BT*DIM
    const float* __restrict__ start_trace,  // (DIM, MTR)
    const float* __restrict__ w1,           // (MTR, 2*HID, DIM)
    const float* __restrict__ b1,           // (1, DIM, 2*HID)
    const float* __restrict__ w2,           // (HID, 2, DIM)
    const float* __restrict__ b2,           // (1, DIM, 2)
    float* __restrict__ state,              // (BT, DIM)
    _Float16* __restrict__ pre,             // (BT, TWO_D)
    int k) {
    int g = blockIdx.x * blockDim.x + threadIdx.x;   // BT*DIM
    if (g >= BT * DIM) return;
    int n = g % DIM;
    int b = g / DIM;

    float t[MTR];
#pragma unroll
    for (int m = 0; m < MTR; ++m) {
        if (m < 7 - k)
            t[m] = start_trace[n * MTR + m + k + 1];
        else
            t[m] = nhist[(size_t)(m + k - 7) * (BT * DIM) + (size_t)b * DIM + n];
    }

    float h[HID];
#pragma unroll
    for (int o = 0; o < HID; ++o) {
        float v  = b1[n * (2 * HID) + o];
        float gt = b1[n * (2 * HID) + o + HID];
#pragma unroll
        for (int m = 0; m < MTR; ++m) {
            v  += t[m] * w1[m * (2 * HID * DIM) + o * DIM + n];
            gt += t[m] * w1[m * (2 * HID * DIM) + (o + HID) * DIM + n];
        }
        h[o] = v * sigmoidf(gt);
    }

    float u0 = b2[n * 2 + 0];
    float u1 = b2[n * 2 + 1];
#pragma unroll
    for (int m = 0; m < HID; ++m) {
        u0 += h[m] * w2[m * (2 * DIM) + n];
        u1 += h[m] * w2[m * (2 * DIM) + DIM + n];
    }
    float st = u0 * sigmoidf(u1);
    state[g] = st;
    pre[(size_t)b * TWO_D + DIM + n] = (_Float16)st;
}

// =====================================================================
// Synch EMA update: alpha = r*alpha + state[l]*state[r]  (k==0: alpha=0)
// =====================================================================
__global__ __launch_bounds__(256) void pp_alpha(
    const float* __restrict__ state, const int* __restrict__ sl,
    const int* __restrict__ sr, const float* __restrict__ decay,
    float* __restrict__ alpha, int k) {
    int g = blockIdx.x * blockDim.x + threadIdx.x;   // BT*NS
    if (g >= BT * NS) return;
    int s = g & (NS - 1);
    int b = g >> 9;
    float r  = expf(-fminf(fmaxf(decay[s], 0.0f), 15.0f));
    float pp = state[(size_t)b * DIM + sl[s]] * state[(size_t)b * DIM + sr[s]];
    float a  = (k == 0) ? 0.0f : alpha[g];
    alpha[g] = r * a + pp;
}

// synch = alpha / sqrt(beta_K), beta_K from closed recurrence; emit f16
__global__ __launch_bounds__(256) void make_synch(
    const float* __restrict__ alpha, const float* __restrict__ decay,
    _Float16* __restrict__ synch) {
    int g = blockIdx.x * blockDim.x + threadIdx.x;   // BT*NS
    if (g >= BT * NS) return;
    int s = g & (NS - 1);
    float r = expf(-fminf(fmaxf(decay[s], 0.0f), 15.0f));
    float beta = 0.0f;
#pragma unroll
    for (int k = 0; k < KIT; ++k) beta = r * beta + 1.0f;
    synch[g] = (_Float16)(alpha[g] * rsqrtf(beta));
}

// =====================================================================
// Host launcher
// =====================================================================
extern "C" void kernel_launch(void* const* d_in, const int* in_sizes, int n_in,
                              void* d_out, int out_size, void* d_ws, size_t ws_size,
                              hipStream_t stream) {
    const float* x        = (const float*)d_in[0];
    const float* W_syn    = (const float*)d_in[1];
    const float* ln_g     = (const float*)d_in[2];
    const float* ln_b     = (const float*)d_in[3];
    const float* w_nlm1   = (const float*)d_in[4];
    const float* b_nlm1   = (const float*)d_in[5];
    const float* w_nlm2   = (const float*)d_in[6];
    const float* b_nlm2   = (const float*)d_in[7];
    const float* start_st = (const float*)d_in[8];
    const float* start_tr = (const float*)d_in[9];
    const float* decay    = (const float*)d_in[10];
    const float* W_proj   = (const float*)d_in[11];
    const int*   sl       = (const int*)d_in[12];
    const int*   sr       = (const int*)d_in[13];
    float* out            = (float*)d_out;

    // ---- workspace carve-up (256B aligned) ----
    char* p = (char*)d_ws;
    auto carve = [&](size_t bytes) {
        char* r = p;
        p += (bytes + 255) & ~(size_t)255;
        return r;
    };
    _Float16* wsyn16  = (_Float16*)carve((size_t)TWO_D * TWO_D * 2);
    _Float16* wproj16 = (_Float16*)carve((size_t)DIM * NS * 2);
    _Float16* pre16   = (_Float16*)carve((size_t)BT * TWO_D * 2);
    float*    act     = (float*)   carve((size_t)BT * TWO_D * 4);
    float*    nhist   = (float*)   carve((size_t)KIT * BT * DIM * 4);
    float*    state   = (float*)   carve((size_t)BT * DIM * 4);
    float*    alpha   = (float*)   carve((size_t)BT * NS * 4);
    _Float16* synch16 = (_Float16*)carve((size_t)BT * NS * 2);

    // ---- prep ----
    {
        int n = TWO_D * TWO_D;
        cvt_f32_f16<<<(n + 255) / 256, 256, 0, stream>>>(W_syn, wsyn16, n);
    }
    {
        int n = DIM * NS;
        cvt_f32_f16<<<(n + 255) / 256, 256, 0, stream>>>(W_proj, wproj16, n);
    }
    init_pre<<<(BT * TWO_D + 255) / 256, 256, 0, stream>>>(x, start_st, pre16);

    // ---- K recurrent steps ----
    for (int k = 0; k < KIT; ++k) {
        wmma_gemm_nt<<<dim3(TWO_D / 128, BT / 256), 256, 0, stream>>>(
            pre16, wsyn16, act, BT, TWO_D, TWO_D);
        glu_ln<<<BT, 256, 0, stream>>>(
            act, ln_g, ln_b, nhist + (size_t)k * BT * DIM);
        nlm_step<<<(BT * DIM + 255) / 256, 256, 0, stream>>>(
            nhist, start_tr, w_nlm1, b_nlm1, w_nlm2, b_nlm2, state, pre16, k);
        pp_alpha<<<(BT * NS + 255) / 256, 256, 0, stream>>>(
            state, sl, sr, decay, alpha, k);
    }

    // ---- close: synch + projection ----
    make_synch<<<(BT * NS + 255) / 256, 256, 0, stream>>>(alpha, decay, synch16);
    wmma_gemm_nt<<<dim3(DIM / 128, BT / 256), 256, 0, stream>>>(
        synch16, wproj16, out, BT, DIM, NS);
}
